// PerformerEncoder_51479478010630
// MI455X (gfx1250) — matrix operand order repaired and checked by reading, not compiled, not run on hardware
//
#include <hip/hip_runtime.h>

#ifndef __has_builtin
#define __has_builtin(x) 0
#endif

#if __has_builtin(__builtin_amdgcn_global_load_async_to_lds_b128)
#define HAVE_ASYNC_LDS 1
#else
#define HAVE_ASYNC_LDS 0
#endif
#if __has_builtin(__builtin_amdgcn_tensor_load_to_lds) && \
    __has_builtin(__builtin_amdgcn_s_wait_tensorcnt)
#define HAVE_TDM 1
#else
#define HAVE_TDM 0
#endif

// ---------- types ----------
typedef __attribute__((ext_vector_type(16))) __bf16   v16bf;
typedef __attribute__((ext_vector_type(8)))  float    v8f;
typedef __attribute__((ext_vector_type(4)))  unsigned int u32x4;
typedef __attribute__((ext_vector_type(8)))  int      i32x8;
typedef __attribute__((ext_vector_type(4)))  int      i32x4;

// vector-size-attribute int4 matching the async-LDS builtin's pointee type
typedef int v4i_vs __attribute__((__vector_size__(16)));
typedef __attribute__((address_space(1))) v4i_vs* as1_v4i_ptr;
typedef __attribute__((address_space(3))) v4i_vs* as3_v4i_ptr;

union FragBF { v16bf v; u32x4 q[2]; };

// ---------- problem constants ----------
constexpr int kB = 4, kS = 4096, kN = 16384;    // tokens
constexpr int kE = 512, kH = 8, kD = 64, kM = 2048, kL = 6;
constexpr int kQKV = 1536;                      // 3*H*D

// ---------- helpers ----------
__device__ __forceinline__ unsigned short f2bf(float f) {
  unsigned int u = __float_as_uint(f);
  unsigned int r = (u + 0x7FFFu + ((u >> 16) & 1u)) >> 16;
  return (unsigned short)r;
}
__device__ __forceinline__ float bf2f(unsigned short s) {
  return __uint_as_float(((unsigned int)s) << 16);
}
__device__ __forceinline__ float gelu_tanh(float x) {
  const float c = 0.7978845608028654f;
  float t = tanhf(c * (x + 0.044715f * x * x * x));
  return 0.5f * x * (1.0f + t);
}
__device__ __forceinline__ v8f v8f_zero() {
  v8f z; for (int i = 0; i < 8; ++i) z[i] = 0.0f; return z;
}

#if HAVE_ASYNC_LDS
__device__ __forceinline__ void async_ld128(const void* g, void* l) {
  __builtin_amdgcn_global_load_async_to_lds_b128(
      (as1_v4i_ptr)g, (as3_v4i_ptr)l, 0, 0);
}
__device__ __forceinline__ void wait_async0() {
#if __has_builtin(__builtin_amdgcn_s_wait_asynccnt)
  __builtin_amdgcn_s_wait_asynccnt(0);
#else
  asm volatile("s_wait_asynccnt 0" ::: "memory");
#endif
}
#endif

// ---------- weight pack: fp32 [L][K][N] -> bf16 [L][N][K] (transposed) ----------
__global__ void pack_transpose_kernel(const float* __restrict__ src,
                                      unsigned short* __restrict__ dst,
                                      int K, int N,
                                      size_t srcLayerStride, size_t dstLayerStride,
                                      size_t dstOffset, size_t total) {
  size_t idx = (size_t)blockIdx.x * 256 + threadIdx.x;
  if (idx >= total) return;
  int k = (int)(idx % K);
  size_t t = idx / K;
  int n = (int)(t % N);
  int l = (int)(t / N);
  dst[(size_t)l * dstLayerStride + dstOffset + (size_t)n * K + k] =
      f2bf(src[(size_t)l * srcLayerStride + (size_t)k * N + n]);
}

// ---------- embedding + sinusoidal positions ----------
__global__ void embed_kernel(const int* __restrict__ inputs,
                             const float* __restrict__ emb,
                             float* __restrict__ x) {
  int n = blockIdx.x;
  int s = n % kS;
  int tok = inputs[n];
  for (int e = threadIdx.x; e < kE; e += 256) {
    int i2 = e & ~1;
    float div = expf((float)i2 * (-9.210340371976184f / (float)kE));
    float ang = (float)s * div;
    float pe = (e & 1) ? cosf(ang) : sinf(ang);
    x[(size_t)n * kE + e] = emb[(size_t)tok * kE + e] + pe;
  }
}

// ---------- layer norm (256 threads, 1 token/block, E=512) ----------
template <bool OUT_BF16>
__global__ __launch_bounds__(256) void ln_kernel(const float* __restrict__ x,
                                                 const float* __restrict__ sc,
                                                 const float* __restrict__ bi,
                                                 unsigned short* __restrict__ ob,
                                                 float* __restrict__ of) {
  __shared__ float red[256];
  int n = blockIdx.x, tid = threadIdx.x;
  const float* xr = x + (size_t)n * kE;
  float v0 = xr[tid], v1 = xr[tid + 256];
  red[tid] = v0 + v1;
  __syncthreads();
  for (int off = 128; off > 0; off >>= 1) {
    if (tid < off) red[tid] += red[tid + off];
    __syncthreads();
  }
  float mu = red[0] * (1.0f / kE);
  __syncthreads();
  float d0 = v0 - mu, d1 = v1 - mu;
  red[tid] = d0 * d0 + d1 * d1;
  __syncthreads();
  for (int off = 128; off > 0; off >>= 1) {
    if (tid < off) red[tid] += red[tid + off];
    __syncthreads();
  }
  float rs = rsqrtf(red[0] * (1.0f / kE) + 1e-6f);
  float o0 = d0 * rs * sc[tid] + bi[tid];
  float o1 = d1 * rs * sc[tid + 256] + bi[tid + 256];
  if (OUT_BF16) {
    ob[(size_t)n * kE + tid] = f2bf(o0);
    ob[(size_t)n * kE + tid + 256] = f2bf(o1);
  } else {
    of[(size_t)n * kE + tid] = o0;
    of[(size_t)n * kE + tid + 256] = o1;
  }
}

// ---------- main WMMA GEMM: C[M,N] = A[M,K](bf16) x Bt[N,K](bf16) ----------
// EPI: 0 = store bf16; 1 = +bias, gelu, store bf16; 2 = resadd into f32 C;
//      3 = +bias, resadd into f32 C; 4 = FAVOR phi epilogue (QKV): cols<512
//      relu+eps, cols 512..1023 (relu+eps)*mask(row), cols>=1024 passthrough.
template <int EPI>
__global__ __launch_bounds__(256) void gemm_kernel(const unsigned short* __restrict__ A,
                                                   const unsigned short* __restrict__ Bt,
                                                   float* __restrict__ Cf,
                                                   unsigned short* __restrict__ Cb,
                                                   const float* __restrict__ bias,
                                                   const int* __restrict__ msk,
                                                   int Mdim, int Ndim, int Kdim) {
  __shared__ __align__(16) unsigned short As[2][128][40];
  __shared__ __align__(16) unsigned short Bs[2][128][40];
  int tid = threadIdx.x;
  int n0 = blockIdx.x * 128, m0 = blockIdx.y * 128;
  int wave = tid >> 5, lane = tid & 31;
  int wm = wave >> 1, wn = wave & 1;   // 4x2 waves: 32 rows x 64 cols each
  int l = lane & 15, hi = lane >> 4;

  v8f acc[2][4];
#pragma unroll
  for (int t = 0; t < 2; ++t)
#pragma unroll
    for (int j = 0; j < 4; ++j) acc[t][j] = v8f_zero();

  int rowA = tid >> 2, chA = (tid & 3) * 8;
  int rowA2 = (tid + 256) >> 2, chA2 = ((tid + 256) & 3) * 8;

  auto issue = [&](int buf, int k0) {
#if HAVE_ASYNC_LDS
    async_ld128(A + (size_t)(m0 + rowA) * Kdim + k0 + chA,   &As[buf][rowA][chA]);
    async_ld128(A + (size_t)(m0 + rowA2) * Kdim + k0 + chA2, &As[buf][rowA2][chA2]);
    async_ld128(Bt + (size_t)(n0 + rowA) * Kdim + k0 + chA,   &Bs[buf][rowA][chA]);
    async_ld128(Bt + (size_t)(n0 + rowA2) * Kdim + k0 + chA2, &Bs[buf][rowA2][chA2]);
#else
    *(u32x4*)&As[buf][rowA][chA]   = *(const u32x4*)(A + (size_t)(m0 + rowA) * Kdim + k0 + chA);
    *(u32x4*)&As[buf][rowA2][chA2] = *(const u32x4*)(A + (size_t)(m0 + rowA2) * Kdim + k0 + chA2);
    *(u32x4*)&Bs[buf][rowA][chA]   = *(const u32x4*)(Bt + (size_t)(n0 + rowA) * Kdim + k0 + chA);
    *(u32x4*)&Bs[buf][rowA2][chA2] = *(const u32x4*)(Bt + (size_t)(n0 + rowA2) * Kdim + k0 + chA2);
    if (k0 + 32 < Kdim) {  // CDNA5 global_prefetch_b8
      __builtin_prefetch(A + (size_t)(m0 + rowA) * Kdim + k0 + 32 + chA, 0, 3);
      __builtin_prefetch(Bt + (size_t)(n0 + rowA) * Kdim + k0 + 32 + chA, 0, 3);
    }
#endif
  };

  int nsteps = Kdim >> 5;
  issue(0, 0);
  for (int step = 0; step < nsteps; ++step) {
    int cur = step & 1;
#if HAVE_ASYNC_LDS
    wait_async0();               // this wave's async LDS writes done
#endif
    __syncthreads();             // tiles visible; prior compute on other buf done
    if (step + 1 < nsteps) issue(1 - cur, (step + 1) * 32);

    FragBF a[2], b[4];
#pragma unroll
    for (int t = 0; t < 2; ++t) {
      const unsigned short* p = &As[cur][wm * 32 + t * 16 + l][hi ? 8 : 0];
      a[t].q[0] = *(const u32x4*)p;
      a[t].q[1] = *(const u32x4*)(p + 16);
    }
#pragma unroll
    for (int j = 0; j < 4; ++j) {
      const unsigned short* p = &Bs[cur][wn * 64 + j * 16 + l][hi * 8];
      b[j].q[0] = *(const u32x4*)p;
      b[j].q[1] = *(const u32x4*)(p + 16);
    }
#pragma unroll
    for (int t = 0; t < 2; ++t)
#pragma unroll
      for (int j = 0; j < 4; ++j)
        acc[t][j] = __builtin_amdgcn_wmma_f32_16x16x32_bf16(
            false, a[t].v, false, b[j].v, (short)0, acc[t][j], false, false);
    __syncthreads();
  }

#pragma unroll
  for (int t = 0; t < 2; ++t) {
    int rbase = m0 + wm * 32 + t * 16 + hi * 8;
#pragma unroll
    for (int j = 0; j < 4; ++j) {
      int col = n0 + wn * 64 + j * 16 + l;
      float bv = (EPI == 1 || EPI == 3) ? bias[col] : 0.0f;
#pragma unroll
      for (int r = 0; r < 8; ++r) {
        size_t off = (size_t)(rbase + r) * Ndim + col;
        float v = acc[t][j][r];
        if (EPI == 0) {
          Cb[off] = f2bf(v);
        } else if (EPI == 1) {
          Cb[off] = f2bf(gelu_tanh(v + bv));
        } else if (EPI == 2) {
          Cf[off] += v;
        } else if (EPI == 3) {
          Cf[off] += v + bv;
        } else {  // EPI == 4: fused FAVOR phi
          if (col >= 1024) {
            Cb[off] = f2bf(v);
          } else {
            float r_ = fmaxf(v, 0.0f) + 1e-3f;
            if (col >= 512 && msk[rbase + r] <= 0) r_ = 0.0f;
            Cb[off] = f2bf(r_);
          }
        }
      }
    }
  }
}

// ---------- z[bh][m] = sum_s phi_k ----------
__global__ void z_kernel(const unsigned short* __restrict__ qkv, float* __restrict__ z) {
  int bh = blockIdx.x;
  int b = bh / kH, h = bh % kH, m = threadIdx.x;  // 64 threads
  const unsigned short* p = qkv + (size_t)b * kS * kQKV + 512 + h * 64 + m;
  float s = 0.0f;
  for (int i = 0; i < kS; ++i) s += bf2f(p[(size_t)i * kQKV]);
  z[bh * 64 + m] = s;
}

// ---------- kvT[bh][d][m] = sum_s phi_k[s,m] * v[s,d] (VALU; ~1% of FLOPs) ----------
__global__ __launch_bounds__(256) void kv_kernel(const unsigned short* __restrict__ qkv,
                                                 unsigned short* __restrict__ kvT) {
  int bh = blockIdx.x;
  int b = bh / kH, h = bh % kH;
  __shared__ __align__(16) unsigned short pk[32][64];
  __shared__ __align__(16) unsigned short vv[32][64];
  int tid = threadIdx.x;
  int mi = (tid >> 4) * 4, di = (tid & 15) * 4;
  float acc[4][4] = {};
  const unsigned short* base = qkv + (size_t)b * kS * kQKV;
  int row = tid >> 3, c8 = (tid & 7) * 8;
  for (int s0 = 0; s0 < kS; s0 += 32) {
    *(u32x4*)&pk[row][c8] = *(const u32x4*)(base + (size_t)(s0 + row) * kQKV + 512 + h * 64 + c8);
    *(u32x4*)&vv[row][c8] = *(const u32x4*)(base + (size_t)(s0 + row) * kQKV + 1024 + h * 64 + c8);
    __syncthreads();
#pragma unroll 4
    for (int ss = 0; ss < 32; ++ss) {
      float pm[4], vd[4];
#pragma unroll
      for (int a = 0; a < 4; ++a) pm[a] = bf2f(pk[ss][mi + a]);
#pragma unroll
      for (int c = 0; c < 4; ++c) vd[c] = bf2f(vv[ss][di + c]);
#pragma unroll
      for (int a = 0; a < 4; ++a)
#pragma unroll
        for (int c = 0; c < 4; ++c) acc[a][c] = fmaf(pm[a], vd[c], acc[a][c]);
    }
    __syncthreads();
  }
  unsigned short* out = kvT + (size_t)bh * 64 * 64;
#pragma unroll
  for (int a = 0; a < 4; ++a)
#pragma unroll
    for (int c = 0; c < 4; ++c)
      out[(di + c) * 64 + (mi + a)] = f2bf(acc[a][c]);  // transposed store
}

// ---------- den[n*H+h] = phi_q . z ----------
__global__ void den_kernel(const unsigned short* __restrict__ qkv,
                           const float* __restrict__ z, float* __restrict__ den) {
  int idx = blockIdx.x * 256 + threadIdx.x;
  if (idx >= kN * kH) return;
  int n = idx / kH, h = idx % kH, b = n / kS;
  const unsigned short* p = qkv + (size_t)n * kQKV + h * 64;
  const float* zz = z + (b * kH + h) * 64;
  float s = 0.0f;
#pragma unroll 8
  for (int m = 0; m < 64; ++m) s = fmaf(bf2f(p[m]), zz[m], s);
  den[idx] = s;
}

// ---------- o = (phi_q @ kv) / den  via WMMA (K=64); kv tile via TDM ----------
__global__ __launch_bounds__(256) void attn_out_kernel(const unsigned short* __restrict__ qkv,
                                                       const unsigned short* __restrict__ kvT,
                                                       const float* __restrict__ den,
                                                       unsigned short* __restrict__ o) {
  int n0 = blockIdx.x * 128;
  int h = blockIdx.y;
  int b = n0 / kS;
  __shared__ __align__(16) unsigned short kvs[64][64];
  int tid = threadIdx.x;
#if HAVE_TDM
  if (tid < 32) {  // wave 0 drives the Tensor Data Mover (EXEC-independent DMA)
    unsigned lds = (unsigned)(unsigned long long)
        (__attribute__((address_space(3))) void*)&kvs[0][0];
    unsigned long long ga =
        (unsigned long long)(kvT + (size_t)(b * kH + h) * 4096);
    u32x4 g0; i32x8 g1; i32x4 g2; i32x4 g3; i32x8 g4;
    g0[0] = 1u;                                   // count=1, user descriptor
    g0[1] = lds;                                  // lds_addr
    g0[2] = (unsigned)ga;                         // global_addr[31:0]
    g0[3] = (unsigned)((ga >> 32) & 0x01FFFFFFu)  // global_addr[56:32]
            | 0x80000000u;                        // type=2 ("image")
    g1[0] = 0x00010000;          // data_size=2B
    g1[1] = (int)(4096u << 16);  // tensor_dim0[15:0]=4096
    g1[2] = (int)(1u << 16);     // tensor_dim1=1
    g1[3] = (int)(4096u << 16);  // tile_dim0=4096
    g1[4] = 1;                   // tile_dim1=1
    g1[5] = 4096;                // tensor_dim0_stride
    g1[6] = 0; g1[7] = 0;
    g2[0] = g2[1] = g2[2] = g2[3] = 0;
    g3[0] = g3[1] = g3[2] = g3[3] = 0;
    for (int i = 0; i < 8; ++i) g4[i] = 0;
    __builtin_amdgcn_tensor_load_to_lds(g0, g1, g2, g3, g4, 0);
    __builtin_amdgcn_s_wait_tensorcnt(0);
  }
  __syncthreads();
#else
  {
    const u32x4* src = (const u32x4*)(kvT + (size_t)(b * kH + h) * 4096);
    ((u32x4*)kvs)[tid] = src[tid];
    ((u32x4*)kvs)[tid + 256] = src[tid + 256];
  }
  __syncthreads();
#endif
  int wave = tid >> 5, lane = tid & 31;
  int l = lane & 15, hi = lane >> 4;
  int row = n0 + wave * 16 + l;

  v8f acc[4];
#pragma unroll
  for (int j = 0; j < 4; ++j) acc[j] = v8f_zero();

#pragma unroll
  for (int ks = 0; ks < 2; ++ks) {
    FragBF a;
    const unsigned short* ap = qkv + (size_t)row * kQKV + h * 64 + ks * 32 + (hi ? 8 : 0);
    a.q[0] = *(const u32x4*)ap;
    a.q[1] = *(const u32x4*)(ap + 16);
#pragma unroll
    for (int j = 0; j < 4; ++j) {
      FragBF bq;
      const unsigned short* bp = &kvs[j * 16 + l][ks * 32 + hi * 8];
      bq.q[0] = *(const u32x4*)bp;
      bq.q[1] = *(const u32x4*)(bp + 16);
      acc[j] = __builtin_amdgcn_wmma_f32_16x16x32_bf16(
          false, a.v, false, bq.v, (short)0, acc[j], false, false);
    }
  }
  float inv[8];
#pragma unroll
  for (int r = 0; r < 8; ++r)
    inv[r] = 1.0f / den[(n0 + wave * 16 + r + hi * 8) * kH + h];
#pragma unroll
  for (int j = 0; j < 4; ++j)
#pragma unroll
    for (int r = 0; r < 8; ++r) {
      int rr = n0 + wave * 16 + r + hi * 8;
      o[(size_t)rr * 512 + h * 64 + j * 16 + l] = f2bf(acc[j][r] * inv[r]);
    }
}

// ---------- workspace layout (bytes) ----------
constexpr size_t OFF_X    = 0;                                  // f32 [N,E]
constexpr size_t OFF_H    = OFF_X    + (size_t)kN * kE * 4;     // bf16 [N,E]
constexpr size_t OFF_BIG  = OFF_H    + (size_t)kN * kE * 2;     // bf16 qkv / ffn-mid
constexpr size_t OFF_O    = OFF_BIG  + (size_t)kN * kM * 2;     // bf16 [N,512]
constexpr size_t OFF_KVT  = OFF_O    + (size_t)kN * 512 * 2;    // bf16 [B*H][64][64]
constexpr size_t OFF_Z    = OFF_KVT  + (size_t)kB * kH * 4096 * 2;
constexpr size_t OFF_DEN  = OFF_Z    + (size_t)kB * kH * 64 * 4;
constexpr size_t OFF_WQKV = OFF_DEN  + (size_t)kN * kH * 4;     // bf16 [L][1536][512]
constexpr size_t OFF_WO   = OFF_WQKV + (size_t)kL * kQKV * kE * 2;
constexpr size_t OFF_W1   = OFF_WO   + (size_t)kL * kE * 512 * 2;
constexpr size_t OFF_W2   = OFF_W1   + (size_t)kL * kM * kE * 2;

extern "C" void kernel_launch(void* const* d_in, const int* in_sizes, int n_in,
                              void* d_out, int out_size, void* d_ws, size_t ws_size,
                              hipStream_t stream) {
  const int*   inputs = (const int*)d_in[0];
  const float* embedw = (const float*)d_in[1];
  const float* ln1s   = (const float*)d_in[2];
  const float* ln1b   = (const float*)d_in[3];
  const float* wq     = (const float*)d_in[4];
  const float* wk     = (const float*)d_in[5];
  const float* wv     = (const float*)d_in[6];
  const float* wo     = (const float*)d_in[7];
  const float* ln2s   = (const float*)d_in[8];
  const float* ln2b   = (const float*)d_in[9];
  const float* w1     = (const float*)d_in[10];
  const float* b1     = (const float*)d_in[11];
  const float* w2     = (const float*)d_in[12];
  const float* b2     = (const float*)d_in[13];
  const float* lnfs   = (const float*)d_in[14];
  const float* lnfb   = (const float*)d_in[15];

  char* ws = (char*)d_ws;
  float*          x    = (float*)(ws + OFF_X);
  unsigned short* hbuf = (unsigned short*)(ws + OFF_H);
  unsigned short* big  = (unsigned short*)(ws + OFF_BIG);   // qkv / ffn-mid
  unsigned short* obuf = (unsigned short*)(ws + OFF_O);
  unsigned short* kvT  = (unsigned short*)(ws + OFF_KVT);
  float*          zbuf = (float*)(ws + OFF_Z);
  float*          den  = (float*)(ws + OFF_DEN);
  unsigned short* wqkvT= (unsigned short*)(ws + OFF_WQKV);
  unsigned short* woT  = (unsigned short*)(ws + OFF_WO);
  unsigned short* w1T  = (unsigned short*)(ws + OFF_W1);
  unsigned short* w2T  = (unsigned short*)(ws + OFF_W2);

  auto blocks = [](size_t total) { return (unsigned)((total + 255) / 256); };

  // ---- pack weights (fp32 -> bf16, transposed to [N][K]) ----
  {
    size_t t;
    t = (size_t)kL * kE * 512;
    pack_transpose_kernel<<<blocks(t), 256, 0, stream>>>(wq, wqkvT, kE, 512,
        (size_t)kE * 512, (size_t)kQKV * kE, 0, t);
    pack_transpose_kernel<<<blocks(t), 256, 0, stream>>>(wk, wqkvT, kE, 512,
        (size_t)kE * 512, (size_t)kQKV * kE, (size_t)512 * kE, t);
    pack_transpose_kernel<<<blocks(t), 256, 0, stream>>>(wv, wqkvT, kE, 512,
        (size_t)kE * 512, (size_t)kQKV * kE, (size_t)1024 * kE, t);
    t = (size_t)kL * 512 * kE;
    pack_transpose_kernel<<<blocks(t), 256, 0, stream>>>(wo, woT, 512, kE,
        (size_t)512 * kE, (size_t)kE * 512, 0, t);
    t = (size_t)kL * kE * kM;
    pack_transpose_kernel<<<blocks(t), 256, 0, stream>>>(w1, w1T, kE, kM,
        (size_t)kE * kM, (size_t)kM * kE, 0, t);
    t = (size_t)kL * kM * kE;
    pack_transpose_kernel<<<blocks(t), 256, 0, stream>>>(w2, w2T, kM, kE,
        (size_t)kM * kE, (size_t)kE * kM, 0, t);
  }

  // ---- embedding + positions ----
  embed_kernel<<<kN, 256, 0, stream>>>(inputs, embedw, x);

  for (int l = 0; l < kL; ++l) {
    // LN1 -> h (bf16)
    ln_kernel<true><<<kN, 256, 0, stream>>>(x, ln1s + l * kE, ln1b + l * kE, hbuf, nullptr);
    // QKV = h @ wqkv with fused phi/mask -> big (bf16 [N,1536])
    gemm_kernel<4><<<dim3(kQKV / 128, kN / 128), 256, 0, stream>>>(
        hbuf, wqkvT + (size_t)l * kQKV * kE, nullptr, big, nullptr, inputs, kN, kQKV, kE);
    // z, kv, den
    z_kernel<<<kB * kH, 64, 0, stream>>>(big, zbuf);
    kv_kernel<<<kB * kH, 256, 0, stream>>>(big, kvT);
    den_kernel<<<blocks((size_t)kN * kH), 256, 0, stream>>>(big, zbuf, den);
    // o = (phi_q @ kv) / den
    attn_out_kernel<<<dim3(kN / 128, kH), 256, 0, stream>>>(big, kvT, den, obuf);
    // x += o @ wo
    gemm_kernel<2><<<dim3(512 / 128, kN / 128), 256, 0, stream>>>(
        obuf, woT + (size_t)l * kE * 512, x, nullptr, nullptr, nullptr, kN, 512, kE);
    // LN2 -> h (bf16, reuse)
    ln_kernel<true><<<kN, 256, 0, stream>>>(x, ln2s + l * kE, ln2b + l * kE, hbuf, nullptr);
    // g = gelu(h @ w1 + b1) -> big (bf16 [N,2048])
    gemm_kernel<1><<<dim3(kM / 128, kN / 128), 256, 0, stream>>>(
        hbuf, w1T + (size_t)l * kM * kE, nullptr, big, b1 + (size_t)l * kM, nullptr, kN, kM, kE);
    // x += g @ w2 + b2
    gemm_kernel<3><<<dim3(512 / 128, kN / 128), 256, 0, stream>>>(
        big, w2T + (size_t)l * kE * kM, x, nullptr, b2 + (size_t)l * kE, nullptr, kN, 512, kM);
  }

  // final LN -> d_out (fp32)
  ln_kernel<false><<<kN, 256, 0, stream>>>(x, lnfs, lnfb, nullptr, (float*)d_out);
}
